// LanguageModel_38920993636643
// MI455X (gfx1250) — matrix operand order repaired and checked by reading, not compiled, not run on hardware
//
#include <hip/hip_runtime.h>
#include <math.h>

typedef __attribute__((ext_vector_type(16))) _Float16 v16h;
typedef __attribute__((ext_vector_type(8)))  float    v8f;

constexpr int cB = 8, cS = 512, cH = 768, cNH = 12, cDH = 64, cL = 12, cV = 21128, cFF = 3072;
constexpr int cMT = cB * cS;   // 4096 tokens
constexpr float cEPS = 1e-12f;

// ---------------- generic strided-batched WMMA GEMM ----------------
// C[m,n] = act( sum_k A[m,k]*B[k,n] + bias[n] ),  A f32 row-major [M,K] (lda)
// B f32: transB==0 -> [K,N] row-major (ldb);  transB==1 -> element (k,n)=B[n*ldb+k]
// batch: z -> zb=z/batch_h, zh=z%batch_h, ptr += zb*s?b + zh*s?h
// Double-buffered LDS, branch-free (select-predicated) staging loads.
#define TBM 128
#define TBN 64
#define TBK 32
#define LPAD 40   // 32 halves + 8 pad (keeps 4B alignment, dodges conflicts)

__global__ __launch_bounds__(256)
void gemm_wmma(const float* __restrict__ A, const float* __restrict__ Bm,
               const float* __restrict__ bias, float* __restrict__ C,
               int M, int N, int K, int lda, int ldb, int ldc,
               int batch_h,
               long long sAb, long long sAh, long long sBb, long long sBh,
               long long sCb, long long sCh,
               int transB, int do_gelu)
{
  __shared__ _Float16 sA[2][TBM * LPAD];
  __shared__ _Float16 sB[2][TBN * LPAD];

  const int z  = blockIdx.z;
  const int zb = z / batch_h, zh = z % batch_h;
  A  += (size_t)zb * sAb + (size_t)zh * sAh;
  Bm += (size_t)zb * sBb + (size_t)zh * sBh;
  C  += (size_t)zb * sCb + (size_t)zh * sCh;

  const int tid  = threadIdx.x;
  const int lane = tid & 31;
  const int wave = tid >> 5;
  const int wr   = wave & 3;   // 0..3 -> rows
  const int wc   = wave >> 2;  // 0..1 -> cols
  const int hi   = (lane >= 16) ? 1 : 0;
  const int l15  = lane & 15;
  const int blockM = blockIdx.y * TBM;
  const int blockN = blockIdx.x * TBN;

  // ---- branch-free staging: clamp address, select-zero out-of-range ----
  auto loadA = [&](int k0, float4 (&ra)[4]) {
#pragma unroll
    for (int i = 0; i < 4; ++i) {
      int fid = tid + i * 256;        // 1024 float4
      int row = fid >> 3;             // 0..127
      int kq  = (fid & 7) << 2;       // 0,4,...,28
      int gm  = blockM + row;
      bool ok = gm < M;
      float4 v = *(const float4*)(A + (size_t)(ok ? gm : 0) * lda + k0 + kq);
      v.x = ok ? v.x : 0.f; v.y = ok ? v.y : 0.f;
      v.z = ok ? v.z : 0.f; v.w = ok ? v.w : 0.f;
      ra[i] = v;
    }
  };
  auto storeA = [&](int buf, const float4 (&ra)[4]) {
#pragma unroll
    for (int i = 0; i < 4; ++i) {
      int fid = tid + i * 256;
      int row = fid >> 3;
      int kq  = (fid & 7) << 2;
      _Float16* d = &sA[buf][row * LPAD + kq];
      d[0] = (_Float16)ra[i].x; d[1] = (_Float16)ra[i].y;
      d[2] = (_Float16)ra[i].z; d[3] = (_Float16)ra[i].w;
    }
  };
  auto loadB = [&](int k0, float4 (&rb)[2]) {
    if (!transB) {
#pragma unroll
      for (int i = 0; i < 2; ++i) {
        int fid = tid + i * 256;      // 512 float4
        int row = fid >> 4;           // k: 0..31
        int nq  = (fid & 15) << 2;    // n: 0,4,...,60
        int gn  = blockN + nq;
        bool ok = gn < N;
        float4 v = *(const float4*)(Bm + (size_t)(k0 + row) * ldb + (ok ? gn : 0));
        v.x = ok ? v.x : 0.f; v.y = ok ? v.y : 0.f;
        v.z = ok ? v.z : 0.f; v.w = ok ? v.w : 0.f;
        rb[i] = v;
      }
    } else {
#pragma unroll
      for (int i = 0; i < 2; ++i) {
        int fid  = tid + i * 256;     // 512 float4
        int nrow = fid >> 3;          // n: 0..63
        int kq   = (fid & 7) << 2;    // k: 0,4,...,28
        int gn   = blockN + nrow;
        bool ok  = gn < N;
        float4 v = *(const float4*)(Bm + (size_t)(ok ? gn : 0) * ldb + k0 + kq);
        v.x = ok ? v.x : 0.f; v.y = ok ? v.y : 0.f;
        v.z = ok ? v.z : 0.f; v.w = ok ? v.w : 0.f;
        rb[i] = v;
      }
    }
  };
  auto storeB = [&](int buf, const float4 (&rb)[2]) {
    if (!transB) {
#pragma unroll
      for (int i = 0; i < 2; ++i) {
        int fid = tid + i * 256;
        int row = fid >> 4;
        int nq  = (fid & 15) << 2;
        sB[buf][(nq + 0) * LPAD + row] = (_Float16)rb[i].x;
        sB[buf][(nq + 1) * LPAD + row] = (_Float16)rb[i].y;
        sB[buf][(nq + 2) * LPAD + row] = (_Float16)rb[i].z;
        sB[buf][(nq + 3) * LPAD + row] = (_Float16)rb[i].w;
      }
    } else {
#pragma unroll
      for (int i = 0; i < 2; ++i) {
        int fid  = tid + i * 256;
        int nrow = fid >> 3;
        int kq   = (fid & 7) << 2;
        _Float16* d = &sB[buf][nrow * LPAD + kq];
        d[0] = (_Float16)rb[i].x; d[1] = (_Float16)rb[i].y;
        d[2] = (_Float16)rb[i].z; d[3] = (_Float16)rb[i].w;
      }
    }
  };

  v8f acc[2][2];
#pragma unroll
  for (int a = 0; a < 2; ++a)
#pragma unroll
    for (int b = 0; b < 2; ++b)
#pragma unroll
      for (int r = 0; r < 8; ++r) acc[a][b][r] = 0.0f;

  float4 ra[4], rb[2];
  loadA(0, ra); loadB(0, rb);
  storeA(0, ra); storeB(0, rb);
  __syncthreads();

  int cur = 0;
  for (int k0 = 0; k0 < K; k0 += TBK) {
    const int kn = k0 + TBK;
    const bool more = kn < K;
    if (more) { loadA(kn, ra); loadB(kn, rb); }  // global loads in flight during WMMA

    // ---- fragments (ISA 7.12.2 layouts) + 4 WMMAs on current buffer ----
    v16h afr[2], bfr[2];
#pragma unroll
    for (int tr = 0; tr < 2; ++tr) {
      int m = wr * 32 + tr * 16 + l15;
      const _Float16* base = &sA[cur][m * LPAD];
#pragma unroll
      for (int j = 0; j < 8; ++j) {
        int k = ((j < 4) ? 2 * j : 8 + 2 * j) + hi * 8;  // A K-map
        afr[tr][2 * j]     = base[k];
        afr[tr][2 * j + 1] = base[k + 1];
      }
    }
#pragma unroll
    for (int tc = 0; tc < 2; ++tc) {
      int n = wc * 32 + tc * 16 + l15;
      const _Float16* base = &sB[cur][n * LPAD];
#pragma unroll
      for (int j = 0; j < 8; ++j) {
        int k = ((j < 4) ? 2 * j : 8 + 2 * j) + hi * 8;  // B K-map
        bfr[tc][2 * j]     = base[k];
        bfr[tc][2 * j + 1] = base[k + 1];
      }
    }
#pragma unroll
    for (int tr = 0; tr < 2; ++tr)
#pragma unroll
      for (int tc = 0; tc < 2; ++tc)
        acc[tr][tc] = __builtin_amdgcn_wmma_f32_16x16x32_f16(
            false, afr[tr], false, bfr[tc], (short)0, acc[tr][tc], false, false);

    if (more) { storeA(cur ^ 1, ra); storeB(cur ^ 1, rb); }
    __syncthreads();
    cur ^= 1;
  }

  // ---- epilogue: bias + optional exact GELU, f32 store ----
#pragma unroll
  for (int tr = 0; tr < 2; ++tr)
#pragma unroll
    for (int tc = 0; tc < 2; ++tc) {
      int n = blockN + wc * 32 + tc * 16 + l15;
      if (n >= N) continue;
      float bv = bias ? bias[n] : 0.0f;
#pragma unroll
      for (int r = 0; r < 8; ++r) {
        int m = blockM + wr * 32 + tr * 16 + hi * 8 + r;
        if (m < M) {
          float cv = acc[tr][tc][r] + bv;
          if (do_gelu) cv = 0.5f * cv * (1.0f + erff(cv * 0.70710678118f));
          C[(size_t)m * ldc + n] = cv;
        }
      }
    }
}

// ---------------- block reductions (blockDim == 256) ----------------
__device__ __forceinline__ float blk_sum(float v, float* sh) {
  int t = threadIdx.x;
  sh[t] = v; __syncthreads();
  for (int st = 128; st > 0; st >>= 1) { if (t < st) sh[t] += sh[t + st]; __syncthreads(); }
  float r = sh[0]; __syncthreads(); return r;
}
__device__ __forceinline__ float blk_max(float v, float* sh) {
  int t = threadIdx.x;
  sh[t] = v; __syncthreads();
  for (int st = 128; st > 0; st >>= 1) { if (t < st) sh[t] = fmaxf(sh[t], sh[t + st]); __syncthreads(); }
  float r = sh[0]; __syncthreads(); return r;
}

// ---------------- embedding + LayerNorm ----------------
__global__ __launch_bounds__(256)
void emb_ln(const int* __restrict__ ids, const float* __restrict__ wemb,
            const float* __restrict__ pemb, const float* __restrict__ temb,
            const float* __restrict__ g, const float* __restrict__ bta,
            float* __restrict__ out)
{
  __shared__ float sh[256];
  int tok = blockIdx.x;
  int s   = tok % cS;
  int id  = ids[tok];
  float x[3]; float sum = 0.f, sq = 0.f;
#pragma unroll
  for (int e = 0; e < 3; ++e) {
    int i = threadIdx.x + e * 256;
    float v = wemb[(size_t)id * cH + i] + pemb[(size_t)s * cH + i] + temb[i];
    x[e] = v; sum += v; sq += v * v;
  }
  float m  = blk_sum(sum, sh) * (1.0f / cH);
  float e2 = blk_sum(sq,  sh) * (1.0f / cH);
  float rstd = rsqrtf(fmaxf(e2 - m * m, 0.f) + cEPS);
#pragma unroll
  for (int e = 0; e < 3; ++e) {
    int i = threadIdx.x + e * 256;
    out[(size_t)tok * cH + i] = (x[e] - m) * rstd * g[i] + bta[i];
  }
}

// ---------------- residual add + LayerNorm ----------------
__global__ __launch_bounds__(256)
void add_ln(const float* __restrict__ res, const float* __restrict__ x,
            const float* __restrict__ g, const float* __restrict__ bta,
            float* __restrict__ out)
{
  __shared__ float sh[256];
  int tok = blockIdx.x;
  float v[3]; float sum = 0.f, sq = 0.f;
#pragma unroll
  for (int e = 0; e < 3; ++e) {
    int i = threadIdx.x + e * 256;
    float t = res[(size_t)tok * cH + i] + x[(size_t)tok * cH + i];
    v[e] = t; sum += t; sq += t * t;
  }
  float m  = blk_sum(sum, sh) * (1.0f / cH);
  float e2 = blk_sum(sq,  sh) * (1.0f / cH);
  float rstd = rsqrtf(fmaxf(e2 - m * m, 0.f) + cEPS);
#pragma unroll
  for (int e = 0; e < 3; ++e) {
    int i = threadIdx.x + e * 256;
    out[(size_t)tok * cH + i] = (v[e] - m) * rstd * g[i] + bta[i];
  }
}

// ---------------- prompt length (count labels[0,:] == -100) ----------------
__global__ void count_prompt(const int* __restrict__ labels, int* __restrict__ pl)
{
  __shared__ int sh[512];
  int t = threadIdx.x;
  sh[t] = (labels[t] == -100) ? 1 : 0;
  __syncthreads();
  for (int st = 256; st > 0; st >>= 1) { if (t < st) sh[t] += sh[t + st]; __syncthreads(); }
  if (t == 0) *pl = sh[0];
}

// ---------------- masked, scaled softmax over scores rows ----------------
__global__ __launch_bounds__(256)
void softmax_mask(float* __restrict__ sc, const int* __restrict__ plen)
{
  __shared__ float sh[256];
  int i  = blockIdx.x;       // query index
  int bh = blockIdx.y;       // b*NH + h
  float* row = sc + ((size_t)bh * cS + i) * cS;
  int P = *plen;
  float v[2]; float mx = -3.0e38f;
#pragma unroll
  for (int e = 0; e < 2; ++e) {
    int j = threadIdx.x + e * 256;
    float val = row[j] * 0.125f + (((j < P) || (j <= i)) ? 0.0f : -1e9f);
    v[e] = val; mx = fmaxf(mx, val);
  }
  mx = blk_max(mx, sh);
  float s = 0.f;
#pragma unroll
  for (int e = 0; e < 2; ++e) { v[e] = expf(v[e] - mx); s += v[e]; }
  s = blk_sum(s, sh);
  float inv = 1.0f / s;
#pragma unroll
  for (int e = 0; e < 2; ++e) {
    int j = threadIdx.x + e * 256;
    row[j] = v[e] * inv;
  }
}

// ---------------- CE: init / chunk merge / final ----------------
__global__ void ce_init(float* m, float* s, float* lab)
{
  int t = blockIdx.x * blockDim.x + threadIdx.x;
  if (t < cMT) { m[t] = -3.0e38f; s[t] = 0.f; lab[t] = 0.f; }
}

__global__ __launch_bounds__(256)
void ce_update(const float* __restrict__ chunk, int cn, int v0,
               const int* __restrict__ labels,
               float* __restrict__ m, float* __restrict__ s, float* __restrict__ lab)
{
  __shared__ float sh[256];
  int tok = blockIdx.x;
  const float* row = chunk + (size_t)tok * cn;
  int label = labels[tok];
  float mx = -3.0e38f;
  for (int j = threadIdx.x; j < cn; j += 256) mx = fmaxf(mx, row[j]);
  mx = blk_max(mx, sh);
  float se = 0.f;
  for (int j = threadIdx.x; j < cn; j += 256) {
    float val = row[j];
    se += expf(val - mx);
    if (v0 + j == label) lab[tok] = val;   // exactly one thread matches
  }
  se = blk_sum(se, sh);
  if (threadIdx.x == 0) {
    float om = m[tok], os = s[tok];
    float nm = fmaxf(om, mx);
    s[tok] = os * expf(om - nm) + se * expf(mx - nm);
    m[tok] = nm;
  }
}

__global__ __launch_bounds__(256)
void ce_final(const float* __restrict__ m, const float* __restrict__ s,
              const float* __restrict__ lab, const int* __restrict__ labels,
              float* __restrict__ out)
{
  __shared__ float sh[256];
  float L = 0.f, Cn = 0.f;
  for (int t = threadIdx.x; t < cMT; t += 256) {
    if (labels[t] != -100) {
      L  += (m[t] + logf(s[t]) - lab[t]);   // nll = logsumexp - logit[label]
      Cn += 1.0f;
    }
  }
  L  = blk_sum(L, sh);
  Cn = blk_sum(Cn, sh);
  if (threadIdx.x == 0) out[0] = (Cn > 0.f) ? (L / Cn) : 0.0f;
}

// ---------------- host orchestration ----------------
extern "C" void kernel_launch(void* const* d_in, const int* in_sizes, int n_in,
                              void* d_out, int out_size, void* d_ws, size_t ws_size,
                              hipStream_t stream)
{
  const int*   input_ids = (const int*)  d_in[0];
  const int*   labels    = (const int*)  d_in[1];
  const float* word_emb  = (const float*)d_in[2];
  const float* pos_emb   = (const float*)d_in[3];
  const float* type_emb  = (const float*)d_in[4];
  const float* emb_g     = (const float*)d_in[5];
  const float* emb_b     = (const float*)d_in[6];
  const float* Wq  = (const float*)d_in[7];  const float* bq  = (const float*)d_in[8];
  const float* Wk  = (const float*)d_in[9];  const float* bk  = (const float*)d_in[10];
  const float* Wv  = (const float*)d_in[11]; const float* bv  = (const float*)d_in[12];
  const float* Wo  = (const float*)d_in[13]; const float* bo  = (const float*)d_in[14];
  const float* g1  = (const float*)d_in[15]; const float* b1  = (const float*)d_in[16];
  const float* W1  = (const float*)d_in[17]; const float* bf1 = (const float*)d_in[18];
  const float* W2  = (const float*)d_in[19]; const float* bf2 = (const float*)d_in[20];
  const float* g2  = (const float*)d_in[21]; const float* b2  = (const float*)d_in[22];
  const float* Wc  = (const float*)d_in[23]; const float* bc  = (const float*)d_in[24];

  // workspace layout (floats); peak ~164 MB
  float* ws  = (float*)d_ws;
  const size_t actN = (size_t)cMT * cH;               // 3.1M floats
  float* h   = ws;
  float* h2  = h  + actN;
  float* t0  = h2 + actN;
  float* t1  = t0 + actN;
  float* t2  = t1 + actN;
  float* big = t2 + actN;                             // scores / ffn-mid / logit chunk
  const size_t bigN = (size_t)cB * cNH * cS * cS;     // 25.2M floats
  float* ce_m = big + bigN;
  float* ce_s = ce_m + cMT;
  float* ce_l = ce_s + cMT;
  int*   pl   = (int*)(ce_l + cMT);

  auto gemm = [&](const float* A, const float* Bm, const float* bias, float* C,
                  int M, int N, int K, int lda, int ldb, int ldc,
                  int batches, int bh,
                  long long sAb, long long sAh, long long sBb, long long sBh,
                  long long sCb, long long sCh, int transB, int gelu) {
    dim3 grid((N + TBN - 1) / TBN, (M + TBM - 1) / TBM, batches);
    gemm_wmma<<<grid, 256, 0, stream>>>(A, Bm, bias, C, M, N, K, lda, ldb, ldc,
                                        bh, sAb, sAh, sBb, sBh, sCb, sCh, transB, gelu);
  };

  count_prompt<<<1, 512, 0, stream>>>(labels, pl);
  emb_ln<<<cMT, 256, 0, stream>>>(input_ids, word_emb, pos_emb, type_emb, emb_g, emb_b, h);

  const long long sSH = (long long)cS * cH;      // per-batch stride in token-major act
  const long long sSS = (long long)cS * cS;      // one score matrix

  for (int l = 0; l < cL; ++l) {
    const float* wq = Wq + (size_t)l * cH * cH;  const float* bql = bq + (size_t)l * cH;
    const float* wk = Wk + (size_t)l * cH * cH;  const float* bkl = bk + (size_t)l * cH;
    const float* wv = Wv + (size_t)l * cH * cH;  const float* bvl = bv + (size_t)l * cH;
    const float* wo = Wo + (size_t)l * cH * cH;  const float* bol = bo + (size_t)l * cH;
    const float* w1 = W1 + (size_t)l * cH * cFF; const float* bw1 = bf1 + (size_t)l * cFF;
    const float* w2 = W2 + (size_t)l * cFF * cH; const float* bw2 = bf2 + (size_t)l * cH;
    const float* g1l = g1 + (size_t)l * cH;      const float* b1l = b1 + (size_t)l * cH;
    const float* g2l = g2 + (size_t)l * cH;      const float* b2l = b2 + (size_t)l * cH;

    // Q,K,V projections: [4096,768] x [768,768]
    gemm(h, wq, bql, t0, cMT, cH, cH, cH, cH, cH, 1, 1, 0,0,0,0,0,0, 0, 0);
    gemm(h, wk, bkl, t1, cMT, cH, cH, cH, cH, cH, 1, 1, 0,0,0,0,0,0, 0, 0);
    gemm(h, wv, bvl, t2, cMT, cH, cH, cH, cH, cH, 1, 1, 0,0,0,0,0,0, 0, 0);
    // scores = Q @ K^T per (b,h): 96 x [512,64]x[64,512]
    gemm(t0, t1, nullptr, big, cS, cS, cDH, cH, cH, cS,
         cB * cNH, cNH, sSH, cDH, sSH, cDH, (long long)cNH * sSS, sSS, 1, 0);
    softmax_mask<<<dim3(cS, cB * cNH), 256, 0, stream>>>(big, pl);
    // ctx = attn @ V: 96 x [512,512]x[512,64] -> t0 (q dead)
    gemm(big, t2, nullptr, t0, cS, cDH, cS, cS, cH, cH,
         cB * cNH, cNH, (long long)cNH * sSS, sSS, sSH, cDH, sSH, cDH, 0, 0);
    // output projection -> t1
    gemm(t0, wo, bol, t1, cMT, cH, cH, cH, cH, cH, 1, 1, 0,0,0,0,0,0, 0, 0);
    add_ln<<<cMT, 256, 0, stream>>>(h, t1, g1l, b1l, h2);
    // FFN: gelu(h2 @ W1 + b) -> big ; big @ W2 + b -> t2
    gemm(h2, w1, bw1, big, cMT, cFF, cH, cH, cFF, cFF, 1, 1, 0,0,0,0,0,0, 0, 1);
    gemm(big, w2, bw2, t2, cMT, cH, cFF, cFF, cH, cH, 1, 1, 0,0,0,0,0,0, 0, 0);
    add_ln<<<cMT, 256, 0, stream>>>(h2, t2, g2l, b2l, h);
  }

  // chunked cross-entropy over V=21128 (avoids 346 MB logits)
  ce_init<<<(cMT + 255) / 256, 256, 0, stream>>>(ce_m, ce_s, ce_l);
  const int CHUNK = 5376;   // multiple of 64; last chunk = 5000 (multiple of 4)
  for (int v0 = 0; v0 < cV; v0 += CHUNK) {
    int cn = (cV - v0 < CHUNK) ? (cV - v0) : CHUNK;
    gemm(h, Wc + v0, bc + v0, big, cMT, cn, cH, cH, cV, cn, 1, 1, 0,0,0,0,0,0, 0, 0);
    ce_update<<<cMT, 256, 0, stream>>>(big, cn, v0, labels, ce_m, ce_s, ce_l);
  }
  ce_final<<<1, 256, 0, stream>>>(ce_m, ce_s, ce_l, labels, (float*)d_out);
}
